// FeedForwardNetwork_20160576488086
// MI455X (gfx1250) — matrix operand order repaired and checked by reading, not compiled, not run on hardware
//
#include <hip/hip_runtime.h>
#include <hip/hip_bf16.h>
#include <stddef.h>

// ---------------------------------------------------------------------------
// Sizes (fixed by the reference)
// ---------------------------------------------------------------------------
#define DM 480            // input / output row width
#define H0 384
#define H1 192
#define H2 96
#define GDIM 288          // H1 + H2

#define INV_M0 0.08838834764831845f   // 1/sqrt(128)
#define INV_M1 0.125f                 // 1/sqrt(64)
#define INV_M2 0.17677669529663687f   // 1/sqrt(32)
#define INV_H0 0.05103103630798287f   // 1/sqrt(384)
#define INV_H1 0.07216878364870323f   // 1/sqrt(192)
#define INV_H2 0.10206207261596577f   // 1/sqrt(96)

typedef __bf16 bf16_t;
typedef __attribute__((ext_vector_type(16))) bf16_t v16bf;
typedef __attribute__((ext_vector_type(8)))  bf16_t v8bf;
typedef __attribute__((ext_vector_type(8)))  float  v8f;

// LDS row pitches (bf16 elements, multiples of 8 for 16B-aligned ds loads)
#define SP 392   // >= 384  (sbuf; vbuf aliases into this region after o0)
#define GP 296   // >= 288
#define VP 200   // >= 192  (pitch of vbuf inside the sbuf region)
#define WAVES 2              // waves per block
#define TILES 2              // 16-row tiles per wave (B-fragment reuse factor)
#define PER_TILE_LDS_ELEMS (16 * (SP + GP))   // 11008 elems = 22016 B per tile

// ---------------------------------------------------------------------------
// Helpers
// ---------------------------------------------------------------------------
__device__ __forceinline__ v8f wmma_bf16(v16bf a, v16bf b, v8f c) {
  return __builtin_amdgcn_wmma_f32_16x16x32_bf16(
      /*neg_a=*/false, a, /*neg_b=*/false, b,
      /*c_mod=*/(short)0, c, /*reuse_a=*/false, /*reuse_b=*/false);
}

// Wait for all outstanding LDS ops in this wave (DS ops are in-order per wave
// on CDNA5, so cross-lane store->load through LDS is safe after this).
__device__ __forceinline__ void wave_lds_fence() {
  asm volatile("s_wait_dscnt 0" ::: "memory");
}

// Single-instruction v_rcp_f32 based sigmoid (avoids IEEE division chain).
__device__ __forceinline__ float fast_sigmoid(float x) {
  return __builtin_amdgcn_rcpf(1.0f + __expf(-x));
}

// A-matrix 16x32 bf16 fragment, built from row-major f32 global memory.
// rowp points at this lane's row (row = lane & 15); kbase already includes
// the +8 sub-offset for lanes 16..31.  `stride` is the element stride in K.
__device__ __forceinline__ v16bf load_a_frag_global(const float* __restrict__ rowp,
                                                    int kbase, int stride) {
  v16bf a;
#pragma unroll
  for (int i = 0; i < 8; ++i) a[i] = (bf16_t)rowp[(kbase + i) * stride];
#pragma unroll
  for (int i = 0; i < 8; ++i) a[8 + i] = (bf16_t)rowp[(kbase + 16 + i) * stride];
  return a;
}

// A fragment from row-major bf16 LDS buffer (two aligned 16-byte ds loads).
__device__ __forceinline__ v16bf load_a_frag_lds(const bf16_t* rowp, int kbase) {
  v8bf lo = *(const v8bf*)(rowp + kbase);
  v8bf hi = *(const v8bf*)(rowp + kbase + 16);
  return __builtin_shufflevector(lo, hi, 0, 1, 2, 3, 4, 5, 6, 7,
                                 8, 9, 10, 11, 12, 13, 14, 15);
}

// B-matrix 32x16 bf16 fragment from the pre-packed workspace: one contiguous
// 32-byte per-lane blob (two global_load_b128).
__device__ __forceinline__ v16bf load_b_frag(const bf16_t* __restrict__ frags,
                                             int kt, int nt, int KT, int lane) {
  return *(const v16bf*)(frags + ((size_t)(nt * KT + kt) * 32 + lane) * 16);
}

// ---------------------------------------------------------------------------
// Weight packing: f32 (K x Ncols, row-major) -> bf16 WMMA B-fragment layout.
// One thread per (fragment, lane): writes 16 bf16 values (lane's blob).
// ---------------------------------------------------------------------------
__global__ void pack_wmma_b(const float* __restrict__ W, bf16_t* __restrict__ dst,
                            int K, int Ncols) {
  const int tid = blockIdx.x * blockDim.x + threadIdx.x;
  const int KT = K >> 5, NT = Ncols >> 4;
  const int total = KT * NT * 32;
  if (tid >= total) return;
  const int lane = tid & 31;
  const int frag = tid >> 5;
  const int kt = frag % KT;
  const int nt = frag / KT;
  const int col = nt * 16 + (lane & 15);
  const int k0 = kt * 32 + (lane >> 4) * 8;    // lanes 0-15: K..K+7,K+16..K+23; 16-31: K+8..K+15,K+24..K+31
  bf16_t* o = dst + (size_t)tid * 16;
#pragma unroll
  for (int i = 0; i < 16; ++i)
    o[i] = (bf16_t)W[(size_t)(k0 + (i & 7) + ((i >> 3) << 4)) * Ncols + col];
}

// ---------------------------------------------------------------------------
// Main fused kernel.  Each wave owns TILES consecutive 16-row tiles (B reuse).
// B fragments are double-buffered across loop iterations so the wait for the
// current fragments overlaps the in-flight loads of the next ones.
// __launch_bounds__(64, 1): allow the full VGPR budget (no spills) — latency
// is hidden by ILP/double-buffering, not by wave count.
// ---------------------------------------------------------------------------
__global__ void __launch_bounds__(WAVES * 32, 1)
ffn_wmma_kernel(const float* __restrict__ node_input,
                const float* __restrict__ node_attr,
                const float* __restrict__ b_s,
                const float* __restrict__ b_g,
                const float* __restrict__ b_o,
                const bf16_t* __restrict__ fW0s,
                const bf16_t* __restrict__ fW0g,
                const bf16_t* __restrict__ fW1,
                const bf16_t* __restrict__ fW2,
                const bf16_t* __restrict__ fV0,
                const bf16_t* __restrict__ fV1,
                const bf16_t* __restrict__ fV2,
                float* __restrict__ out, int n) {
  const int lane = threadIdx.x & 31;
  const int wave = threadIdx.x >> 5;
  const int tp = blockIdx.x * WAVES + wave;       // tile-pair index
  const int r0 = tp * 16 * TILES;
  if (r0 >= n) return;                            // n divides 16*TILES exactly

  extern __shared__ char smem_raw[];
  bf16_t* sbuf[TILES];
  bf16_t* gbuf[TILES];
  bf16_t* vbuf[TILES];
#pragma unroll
  for (int t = 0; t < TILES; ++t) {
    bf16_t* base = (bf16_t*)smem_raw +
                   (size_t)(wave * TILES + t) * PER_TILE_LDS_ELEMS;
    sbuf[t] = base;                 // 16 x SP  (silu(s), bf16); dead after o0
    gbuf[t] = base + 16 * SP;       // 16 x GP  (sigmoid gate, bf16)
    vbuf[t] = base;                 // 16 x VP  aliases sbuf region (after o0)
  }

  const int colq = lane & 15;              // B/C/D column (and A row) within tile
  const int hi8 = (lane & 16) ? 8 : 0;     // accumulator row sub-offset
  const int aoff = (lane & 16) ? 8 : 0;    // A-fragment K sub-offset

  const float* arow[TILES];
  float yv[TILES][8];
#pragma unroll
  for (int t = 0; t < TILES; ++t) {
    arow[t] = node_input + (size_t)(r0 + t * 16 + colq) * DM;
#pragma unroll
    for (int j = 0; j < 8; ++j) yv[t][j] = node_attr[r0 + t * 16 + hi8 + j];
  }

  // ---- x0 A-fragments (16 x 128 -> 4 K-steps per tile), reused for S and G
  v16bf a0[TILES][4];
#pragma unroll
  for (int t = 0; t < TILES; ++t)
#pragma unroll
    for (int kt = 0; kt < 4; ++kt)
      a0[t][kt] = load_a_frag_global(arow[t], kt * 32 + aoff, 1);

  // ---- S = silu(x0 @ W0s * y*INV_M0 + b_s) -> sbuf  (B double-buffered) ----
  {
    v16bf bcur[4];
#pragma unroll
    for (int kt = 0; kt < 4; ++kt) bcur[kt] = load_b_frag(fW0s, kt, 0, 4, lane);
    for (int nt = 0; nt < 24; ++nt) {
      v16bf bnxt[4];
      const int ntn = (nt + 1 < 24) ? nt + 1 : nt;
#pragma unroll
      for (int kt = 0; kt < 4; ++kt) bnxt[kt] = load_b_frag(fW0s, kt, ntn, 4, lane);
      v8f acc[TILES] = {};
#pragma unroll
      for (int kt = 0; kt < 4; ++kt)
#pragma unroll
        for (int t = 0; t < TILES; ++t)
          acc[t] = wmma_bf16(a0[t][kt], bcur[kt], acc[t]);
      const int col = nt * 16 + colq;
      const float bias = b_s[col];
#pragma unroll
      for (int t = 0; t < TILES; ++t)
#pragma unroll
        for (int j = 0; j < 8; ++j) {
          float x = acc[t][j] * (yv[t][j] * INV_M0) + bias;
          sbuf[t][(hi8 + j) * SP + col] = (bf16_t)(x * fast_sigmoid(x));
        }
#pragma unroll
      for (int kt = 0; kt < 4; ++kt) bcur[kt] = bnxt[kt];
    }
  }

  // ---- G = sigmoid(x0 @ W0g * y*INV_M0 + b_g) -> gbuf  (B double-buffered) -
  {
    v16bf bcur[4];
#pragma unroll
    for (int kt = 0; kt < 4; ++kt) bcur[kt] = load_b_frag(fW0g, kt, 0, 4, lane);
    for (int nt = 0; nt < 18; ++nt) {
      v16bf bnxt[4];
      const int ntn = (nt + 1 < 18) ? nt + 1 : nt;
#pragma unroll
      for (int kt = 0; kt < 4; ++kt) bnxt[kt] = load_b_frag(fW0g, kt, ntn, 4, lane);
      v8f acc[TILES] = {};
#pragma unroll
      for (int kt = 0; kt < 4; ++kt)
#pragma unroll
        for (int t = 0; t < TILES; ++t)
          acc[t] = wmma_bf16(a0[t][kt], bcur[kt], acc[t]);
      const int col = nt * 16 + colq;
      const float bias = b_g[col];
#pragma unroll
      for (int t = 0; t < TILES; ++t)
#pragma unroll
        for (int j = 0; j < 8; ++j) {
          float x = acc[t][j] * (yv[t][j] * INV_M0) + bias;
          gbuf[t][(hi8 + j) * GP + col] = (bf16_t)fast_sigmoid(x);
        }
#pragma unroll
      for (int kt = 0; kt < 4; ++kt) bcur[kt] = bnxt[kt];
    }
  }
  wave_lds_fence();

  // ---- o0 = S @ V0 * y*INV_H0 + b_o  (acc-resident; B double-buffered) -----
  {
    v8f acc[TILES][8] = {};
    v16bf bcur[8];
#pragma unroll
    for (int nt = 0; nt < 8; ++nt) bcur[nt] = load_b_frag(fV0, 0, nt, 12, lane);
    for (int kt = 0; kt < 12; ++kt) {
      v16bf bnxt[8];
      const int ktn = (kt + 1 < 12) ? kt + 1 : kt;
#pragma unroll
      for (int nt = 0; nt < 8; ++nt) bnxt[nt] = load_b_frag(fV0, ktn, nt, 12, lane);
      v16bf at[TILES];
#pragma unroll
      for (int t = 0; t < TILES; ++t)
        at[t] = load_a_frag_lds(sbuf[t] + colq * SP, kt * 32 + aoff);
#pragma unroll
      for (int nt = 0; nt < 8; ++nt)
#pragma unroll
        for (int t = 0; t < TILES; ++t)
          acc[t][nt] = wmma_bf16(at[t], bcur[nt], acc[t][nt]);
#pragma unroll
      for (int nt = 0; nt < 8; ++nt) bcur[nt] = bnxt[nt];
    }
#pragma unroll
    for (int nt = 0; nt < 8; ++nt) {
      const int col = nt * 16 + colq;
      const float bias = b_o[col];
#pragma unroll
      for (int t = 0; t < TILES; ++t)
#pragma unroll
        for (int j = 0; j < 8; ++j)
          out[(size_t)(r0 + t * 16 + hi8 + j) * DM + col] =
              acc[t][nt][j] * (yv[t][j] * INV_H0) + bias;
    }
  }
  wave_lds_fence();   // sbuf reads complete before vbuf (alias) is written

  // ---- v1 / o1: 3 interleaved channels, K-stride 3 in node_input -----------
  for (int m = 0; m < 3; ++m) {
    v16bf a1[TILES][2];
#pragma unroll
    for (int t = 0; t < TILES; ++t)
#pragma unroll
      for (int kt = 0; kt < 2; ++kt)
        a1[t][kt] = load_a_frag_global(arow[t] + 128 + m, kt * 32 + aoff, 3);
    for (int nt = 0; nt < 12; ++nt) {
      v16bf b[2];
#pragma unroll
      for (int kt = 0; kt < 2; ++kt) b[kt] = load_b_frag(fW1, kt, nt, 2, lane);
      v8f acc[TILES] = {};
#pragma unroll
      for (int kt = 0; kt < 2; ++kt)
#pragma unroll
        for (int t = 0; t < TILES; ++t) acc[t] = wmma_bf16(a1[t][kt], b[kt], acc[t]);
      const int col = nt * 16 + colq;
#pragma unroll
      for (int t = 0; t < TILES; ++t)
#pragma unroll
        for (int j = 0; j < 8; ++j) {
          float gv = (float)gbuf[t][(hi8 + j) * GP + col];
          vbuf[t][(hi8 + j) * VP + col] =
              (bf16_t)(acc[t][j] * (yv[t][j] * INV_M1) * gv);
        }
    }
    wave_lds_fence();

    v8f acc[TILES][4] = {};
    for (int kt = 0; kt < 6; ++kt) {
      v16bf b[4];
#pragma unroll
      for (int nt = 0; nt < 4; ++nt) b[nt] = load_b_frag(fV1, kt, nt, 6, lane);
      v16bf at[TILES];
#pragma unroll
      for (int t = 0; t < TILES; ++t)
        at[t] = load_a_frag_lds(vbuf[t] + colq * VP, kt * 32 + aoff);
#pragma unroll
      for (int nt = 0; nt < 4; ++nt)
#pragma unroll
        for (int t = 0; t < TILES; ++t)
          acc[t][nt] = wmma_bf16(at[t], b[nt], acc[t][nt]);
    }
#pragma unroll
    for (int nt = 0; nt < 4; ++nt) {
      const int w = nt * 16 + colq;
#pragma unroll
      for (int t = 0; t < TILES; ++t)
#pragma unroll
        for (int j = 0; j < 8; ++j)
          out[(size_t)(r0 + t * 16 + hi8 + j) * DM + 128 + w * 3 + m] =
              acc[t][nt][j] * (yv[t][j] * INV_H1);
    }
    wave_lds_fence();
  }

  // ---- v2 / o2: 5 interleaved channels, K-stride 5 -------------------------
  for (int m = 0; m < 5; ++m) {
    v16bf a2[TILES];
#pragma unroll
    for (int t = 0; t < TILES; ++t)
      a2[t] = load_a_frag_global(arow[t] + 320 + m, aoff, 5);
    for (int nt = 0; nt < 6; ++nt) {
      v16bf b = load_b_frag(fW2, 0, nt, 1, lane);
      v8f acc[TILES] = {};
#pragma unroll
      for (int t = 0; t < TILES; ++t) acc[t] = wmma_bf16(a2[t], b, acc[t]);
      const int col = nt * 16 + colq;
#pragma unroll
      for (int t = 0; t < TILES; ++t)
#pragma unroll
        for (int j = 0; j < 8; ++j) {
          float gv = (float)gbuf[t][(hi8 + j) * GP + (H1 + col)];
          vbuf[t][(hi8 + j) * VP + col] =
              (bf16_t)(acc[t][j] * (yv[t][j] * INV_M2) * gv);
        }
    }
    wave_lds_fence();

    v8f acc[TILES][2] = {};
    for (int kt = 0; kt < 3; ++kt) {
      v16bf b[2];
#pragma unroll
      for (int nt = 0; nt < 2; ++nt) b[nt] = load_b_frag(fV2, kt, nt, 3, lane);
      v16bf at[TILES];
#pragma unroll
      for (int t = 0; t < TILES; ++t)
        at[t] = load_a_frag_lds(vbuf[t] + colq * VP, kt * 32 + aoff);
#pragma unroll
      for (int nt = 0; nt < 2; ++nt)
#pragma unroll
        for (int t = 0; t < TILES; ++t)
          acc[t][nt] = wmma_bf16(at[t], b[nt], acc[t][nt]);
    }
#pragma unroll
    for (int nt = 0; nt < 2; ++nt) {
      const int w = nt * 16 + colq;
#pragma unroll
      for (int t = 0; t < TILES; ++t)
#pragma unroll
        for (int j = 0; j < 8; ++j)
          out[(size_t)(r0 + t * 16 + hi8 + j) * DM + 320 + w * 5 + m] =
              acc[t][nt][j] * (yv[t][j] * INV_H2);
    }
    wave_lds_fence();
  }
}

// ---------------------------------------------------------------------------
// Launch
// ---------------------------------------------------------------------------
extern "C" void kernel_launch(void* const* d_in, const int* in_sizes, int n_in,
                              void* d_out, int out_size, void* d_ws, size_t ws_size,
                              hipStream_t stream) {
  const float* node_input = (const float*)d_in[0];
  const float* node_attr  = (const float*)d_in[1];
  const float* W0s = (const float*)d_in[2];
  const float* b_s = (const float*)d_in[3];
  const float* W0g = (const float*)d_in[4];
  const float* b_g = (const float*)d_in[5];
  const float* W1  = (const float*)d_in[6];
  const float* W2  = (const float*)d_in[7];
  const float* V0  = (const float*)d_in[8];
  const float* b_o = (const float*)d_in[9];
  const float* V1  = (const float*)d_in[10];
  const float* V2  = (const float*)d_in[11];
  float* out = (float*)d_out;
  const int n = in_sizes[0] / DM;

  // Workspace layout: packed bf16 B-fragments (total 331,776 bytes).
  bf16_t* ws   = (bf16_t*)d_ws;
  bf16_t* fW0s = ws;                  // 128x384 -> 49152 elems
  bf16_t* fW0g = fW0s + 49152;        // 128x288 -> 36864
  bf16_t* fW1  = fW0g + 36864;        //  64x192 -> 12288
  bf16_t* fW2  = fW1  + 12288;        //  32x96  ->  3072
  bf16_t* fV0  = fW2  + 3072;         // 384x128 -> 49152
  bf16_t* fV1  = fV0  + 49152;        // 192x64  -> 12288
  bf16_t* fV2  = fV1  + 12288;        //  96x32  ->  3072

  struct Pk { const float* w; bf16_t* d; int K, Nc; };
  const Pk pk[7] = {
      {W0s, fW0s, 128, 384}, {W0g, fW0g, 128, 288}, {W1, fW1, 64, 192},
      {W2, fW2, 32, 96},     {V0, fV0, 384, 128},   {V1, fV1, 192, 64},
      {V2, fV2, 96, 32}};
  for (int i = 0; i < 7; ++i) {
    int total = (pk[i].K >> 5) * (pk[i].Nc >> 4) * 32;
    pack_wmma_b<<<(total + 255) / 256, 256, 0, stream>>>(pk[i].w, pk[i].d,
                                                         pk[i].K, pk[i].Nc);
  }

  const int TP = (n + 16 * TILES - 1) / (16 * TILES);   // 6250 tile-pairs
  dim3 grid((TP + WAVES - 1) / WAVES);                  // 3125 blocks
  dim3 block(WAVES * 32);                               // 2 waves
  size_t shmem =
      (size_t)WAVES * TILES * PER_TILE_LDS_ELEMS * sizeof(bf16_t);  // ~88 KB
  ffn_wmma_kernel<<<grid, block, shmem, stream>>>(
      node_input, node_attr, b_s, b_g, b_o,
      fW0s, fW0g, fW1, fW2, fV0, fV1, fV2, out, n);
}
